// BinaryLinear_541165879835
// MI455X (gfx1250) — matrix operand order repaired and checked by reading, not compiled, not run on hardware
//
#include <hip/hip_runtime.h>

// Problem: out[8192,4096] = x[8192,4096] @ (sign(W)*alpha)^T, W[4096,4096], alpha=mean|W|
// Strategy: f16 split-precision WMMA GEMM (x = xh + xl in f16; sign(W) exact in f16).
// Wave tile 64x64 (4x4 WMMA tiles, 2 splits): 12 B/lane per WMMA fetch demand.
// Workspace layout (needs ~161 MB):
//   [0]        float  alpha_sum (sum |W|)
//   [64]       float  partial[1024]
//   [8192]     f16    S  [4096*4096]  (sign(W), row-major [o,i])  32 MiB
//   [+32MiB]   f16    XH [8192*4096]                               64 MiB
//   [+96MiB]   f16    XL [8192*4096]                               64 MiB

typedef _Float16 v16h __attribute__((ext_vector_type(16)));
typedef _Float16 v8h  __attribute__((ext_vector_type(8)));
typedef _Float16 v4h  __attribute__((ext_vector_type(4)));
typedef float    v8f  __attribute__((ext_vector_type(8)));

#define MDIM 8192
#define NDIM 4096
#define KDIM 4096

// ---------- pass 1a: per-block partial sums of |W| ----------
__global__ __launch_bounds__(256)
void reduce_abs_partial(const float* __restrict__ w, float* __restrict__ partial, int n4) {
    __shared__ float sm[256];
    float s = 0.f;
    for (int i = blockIdx.x * blockDim.x + threadIdx.x; i < n4; i += gridDim.x * blockDim.x) {
        float4 v = ((const float4*)w)[i];
        s += fabsf(v.x) + fabsf(v.y) + fabsf(v.z) + fabsf(v.w);
    }
    sm[threadIdx.x] = s;
    __syncthreads();
    for (int off = 128; off > 0; off >>= 1) {
        if ((int)threadIdx.x < off) sm[threadIdx.x] += sm[threadIdx.x + off];
        __syncthreads();
    }
    if (threadIdx.x == 0) partial[blockIdx.x] = sm[0];
}

// ---------- pass 1b: deterministic final reduction ----------
__global__ __launch_bounds__(256)
void reduce_final(const float* __restrict__ partial, float* __restrict__ alpha_sum) {
    __shared__ float sm[256];
    float s = 0.f;
    for (int i = threadIdx.x; i < 1024; i += 256) s += partial[i];
    sm[threadIdx.x] = s;
    __syncthreads();
    for (int off = 128; off > 0; off >>= 1) {
        if ((int)threadIdx.x < off) sm[threadIdx.x] += sm[threadIdx.x + off];
        __syncthreads();
    }
    if (threadIdx.x == 0) alpha_sum[0] = sm[0];
}

// ---------- pass 2: S = sign(W) as f16 (exact: +1 / -1 / 0) ----------
__global__ __launch_bounds__(256)
void binarize_w(const float* __restrict__ w, _Float16* __restrict__ s, int n4) {
    int i = blockIdx.x * blockDim.x + threadIdx.x;
    if (i >= n4) return;
    float4 v = ((const float4*)w)[i];
    v4h r;
    r.x = (_Float16)((v.x > 0.f) ? 1.f : ((v.x < 0.f) ? -1.f : 0.f));
    r.y = (_Float16)((v.y > 0.f) ? 1.f : ((v.y < 0.f) ? -1.f : 0.f));
    r.z = (_Float16)((v.z > 0.f) ? 1.f : ((v.z < 0.f) ? -1.f : 0.f));
    r.w = (_Float16)((v.w > 0.f) ? 1.f : ((v.w < 0.f) ? -1.f : 0.f));
    *(v4h*)(s + 4 * (size_t)i) = r;
}

// ---------- pass 3: split x into f16 hi/lo (xh + xl ~= x, ~22 mantissa bits) ----------
__global__ __launch_bounds__(256)
void split_x(const float* __restrict__ x, _Float16* __restrict__ xh,
             _Float16* __restrict__ xl, int n4) {
    int i = blockIdx.x * blockDim.x + threadIdx.x;
    if (i >= n4) return;
    float4 v = ((const float4*)x)[i];
    v4h h, l;
    h.x = (_Float16)v.x; l.x = (_Float16)(v.x - (float)h.x);
    h.y = (_Float16)v.y; l.y = (_Float16)(v.y - (float)h.y);
    h.z = (_Float16)v.z; l.z = (_Float16)(v.z - (float)h.z);
    h.w = (_Float16)v.w; l.w = (_Float16)(v.w - (float)h.w);
    *(v4h*)(xh + 4 * (size_t)i) = h;
    *(v4h*)(xl + 4 * (size_t)i) = l;
}

// ---------- pass 4: WMMA GEMM ----------
// Block = 256 threads = 8 waves (wave32). Wave tile: 64(M) x 64(N) = 4x4 WMMA tiles.
// Block: 2 waves along M x 4 along N -> block tile 128(M) x 256(N). K-step = 32.
// A fragment (16x32 f16): lane L, row m = L%16; halves e=0..15 map to
//   K = e + 16*(L/16)  -> two contiguous 8-half (16B) chunks.
// B fragment (32x16 f16): lane L, col n = L%16; 16 contiguous K values starting
//   at 16*(L/16). Since B[k,n] = S[n,k], this is a contiguous 32B row read of S.
// C/D (16x16 f32): VGPR r -> m = r + 8*(L/16), n = L%16.
// Tile strides (16 rows * 4096 halves * 2B = 128KiB) fold into the 24-bit
// instruction offset, so only one base pointer per operand stream is needed.
__global__ __launch_bounds__(256)
void gemm_wmma_f16split(const _Float16* __restrict__ xh,
                        const _Float16* __restrict__ xl,
                        const _Float16* __restrict__ s,
                        const float* __restrict__ alpha_sum,
                        float* __restrict__ out) {
    const int lane = threadIdx.x & 31;
    const int wave = threadIdx.x >> 5;
    const int wm = wave & 1;            // 0..1 along M
    const int wn = wave >> 1;           // 0..3 along N
    const int l16 = lane & 15;
    const int lhi = lane >> 4;          // 0/1

    const int m0 = blockIdx.y * 128 + wm * 64;   // wave M base (4 tiles of 16)
    const int n0 = blockIdx.x * 256 + wn * 64;   // wave N base (4 tiles of 16)

    // One base pointer per stream; mt/nt tile strides are compile-time offsets.
    const _Float16* ah = xh + (size_t)(m0 + l16) * KDIM + 16 * lhi;   // + mt*16*KDIM
    const _Float16* al = xl + (size_t)(m0 + l16) * KDIM + 16 * lhi;
    const _Float16* bb = s  + (size_t)(n0 + l16) * KDIM + 16 * lhi;  // + nt*16*KDIM

    v8f acc[4][4];
#pragma unroll
    for (int mt = 0; mt < 4; ++mt)
#pragma unroll
        for (int nt = 0; nt < 4; ++nt)
            acc[mt][nt] = (v8f){0.f,0.f,0.f,0.f,0.f,0.f,0.f,0.f};

    for (int kk = 0; kk < KDIM; kk += 32) {
        // A fragments: 4 m-tiles x {hi,lo} split, two 16B chunks each.
        v16h a[4][2];
#pragma unroll
        for (int mt = 0; mt < 4; ++mt) {
            const size_t t = (size_t)mt * 16 * KDIM + kk;
            v8h h0 = *(const v8h*)(ah + t);
            v8h h1 = *(const v8h*)(ah + t + 8);
            v8h l0 = *(const v8h*)(al + t);
            v8h l1 = *(const v8h*)(al + t + 8);
            a[mt][0] = __builtin_shufflevector(h0, h1, 0,1,2,3,4,5,6,7,8,9,10,11,12,13,14,15);
            a[mt][1] = __builtin_shufflevector(l0, l1, 0,1,2,3,4,5,6,7,8,9,10,11,12,13,14,15);
        }
        // For each B tile: one 32B contiguous read, reused by 8 WMMAs.
#pragma unroll
        for (int nt = 0; nt < 4; ++nt) {
            v16h b = *(const v16h*)(bb + (size_t)nt * 16 * KDIM + kk);
#pragma unroll
            for (int mt = 0; mt < 4; ++mt) {
                acc[mt][nt] = __builtin_amdgcn_wmma_f32_16x16x32_f16(
                    false, a[mt][0], false, b, (short)0, acc[mt][nt], false, false);
                acc[mt][nt] = __builtin_amdgcn_wmma_f32_16x16x32_f16(
                    false, a[mt][1], false, b, (short)0, acc[mt][nt], false, false);
            }
        }
    }

    const float alpha = alpha_sum[0] * (1.0f / ((float)NDIM * (float)KDIM));
#pragma unroll
    for (int mt = 0; mt < 4; ++mt) {
#pragma unroll
        for (int nt = 0; nt < 4; ++nt) {
            const int n = n0 + nt * 16 + l16;
#pragma unroll
            for (int r = 0; r < 8; ++r) {
                const int m = m0 + mt * 16 + r + 8 * lhi;
                // write-once stream: non-temporal, keep L2 for S / x-splits
                __builtin_nontemporal_store(acc[mt][nt][r] * alpha,
                                            &out[(size_t)m * NDIM + n]);
            }
        }
    }
}

extern "C" void kernel_launch(void* const* d_in, const int* in_sizes, int n_in,
                              void* d_out, int out_size, void* d_ws, size_t ws_size,
                              hipStream_t stream) {
    const float* x = (const float*)d_in[0];   // [8192, 4096] f32
    const float* w = (const float*)d_in[1];   // [4096, 4096] f32
    float* out = (float*)d_out;               // [8192, 4096] f32

    char* ws = (char*)d_ws;
    float*    alpha_sum = (float*)(ws + 0);
    float*    partial   = (float*)(ws + 64);
    _Float16* S  = (_Float16*)(ws + 8192);
    _Float16* XH = (_Float16*)(ws + 8192 + (size_t)32 * 1024 * 1024);
    _Float16* XL = (_Float16*)(ws + 8192 + (size_t)96 * 1024 * 1024);

    const int nW4 = (NDIM * KDIM) / 4;        // 4.19M float4
    const int nX4 = (MDIM * KDIM) / 4;        // 8.39M float4

    reduce_abs_partial<<<1024, 256, 0, stream>>>(w, partial, nW4);
    reduce_final<<<1, 256, 0, stream>>>(partial, alpha_sum);
    binarize_w<<<(nW4 + 255) / 256, 256, 0, stream>>>(w, S, nW4);
    split_x<<<(nX4 + 255) / 256, 256, 0, stream>>>(x, XH, XL, nX4);

    dim3 grid(NDIM / 256, MDIM / 128);        // 16 x 64 blocks, 256 thr (8 waves) each
    gemm_wmma_f16split<<<grid, 256, 0, stream>>>(XH, XL, S, alpha_sum, out);
}
